// ATSP_Decoder_38405597561200
// MI455X (gfx1250) — matrix operand order repaired and checked by reading, not compiled, not run on hardware
//
#include <hip/hip_runtime.h>
#include <hip/hip_bf16.h>

// ---------------------------------------------------------------------------
// Problem constants
// ---------------------------------------------------------------------------
#define BB 32
#define NN 500
#define MM 500
#define DD 256
#define NP 512   // padded N
#define MP 512   // padded M
#define FLT_MAX_ 3.402823466e+38f

typedef __attribute__((ext_vector_type(16))) __bf16 v16bf;
typedef __attribute__((ext_vector_type(8)))  float  v8f;

#define DEV __device__ __forceinline__

// ---------------------------------------------------------------------------
// WMMA helper: D = A(16x32 bf16) x B(32x16 bf16) + C(16x16 f32)
// ---------------------------------------------------------------------------
DEV v8f wmma_bf16(v16bf a, v16bf b, v8f c) {
    return __builtin_amdgcn_wmma_f32_16x16x32_bf16(
        /*neg_a=*/false, a, /*neg_b=*/false, b,
        /*c_mod=*/(short)0, c, /*reuse_a=*/false, /*reuse_b=*/false);
}

// ---------------------------------------------------------------------------
// Fragment loaders (wave32), contiguous K order k = 16*grp + s used alike for A and B.
// A 16x32 (16-bit): lanes 0-15 row M=l, K=0..15; lanes 16-31 same
// rows, K=16..31; VGPR v holds K pair 2v,2v+1 within group.
// B 32x16 (16-bit): lanes 0-15 col N=l, K=0..15; lanes 16-31 col N=l-16,
// K=16..31; VGPR v holds K pair 2v,2v+1 within group.
// ---------------------------------------------------------------------------
DEV v16bf load_a_f32(const float* __restrict__ base, int r0, int rmax, int ld,
                     int c0, int lane) {
    int g = lane >> 4, mr = lane & 15;
    int r = r0 + mr; if (r > rmax) r = rmax;           // row guard (clamp)
    const float* p = base + (size_t)r * ld + c0 + g * 16;
    v16bf a;
#pragma unroll
    for (int v = 0; v < 8; ++v) {
        int k0 = 2 * v;
        float2 f = *(const float2*)(p + k0);
        a[2 * v]     = (__bf16)f.x;
        a[2 * v + 1] = (__bf16)f.y;
    }
    return a;
}

DEV v16bf load_a_bf16(const __bf16* __restrict__ base, int r0, int ld, int c0,
                      int lane) {
    int g = lane >> 4, mr = lane & 15;
    const __bf16* p = base + (size_t)(r0 + mr) * ld + c0 + g * 16;
    v16bf a;
#pragma unroll
    for (int v = 0; v < 8; ++v) {
        int k0 = 2 * v;
        a[2 * v]     = p[k0];
        a[2 * v + 1] = p[k0 + 1];
    }
    return a;
}

// B[kk][n] = src[(n0+n)*ld + k0 + kk]   (i.e. src holds B transposed,
// contiguous in K per lane -> coalesced loads)
DEV v16bf load_b_f32(const float* __restrict__ base, int n0, int ld, int k0,
                     int lane) {
    int kg = lane >> 4, n = lane & 15;
    const float* p = base + (size_t)(n0 + n) * ld + k0 + kg * 16;
    v16bf b;
#pragma unroll
    for (int v = 0; v < 8; ++v) {
        float2 f = *(const float2*)(p + 2 * v);
        b[2 * v]     = (__bf16)f.x;
        b[2 * v + 1] = (__bf16)f.y;
    }
    return b;
}

DEV v16bf load_b_bf16(const __bf16* __restrict__ base, int n0, int ld, int k0,
                      int lane) {
    int kg = lane >> 4, n = lane & 15;
    const __bf16* p = base + (size_t)(n0 + n) * ld + k0 + kg * 16;
    v16bf b;
#pragma unroll
    for (int v = 0; v < 8; ++v) {
        b[2 * v]     = p[2 * v];
        b[2 * v + 1] = p[2 * v + 1];
    }
    return b;
}

// ---------------------------------------------------------------------------
// Async copy of one 16x32 bf16 chunk (1 KiB) of the jobs tile into LDS:
// two global_load_async_to_lds_b128 (32 lanes x 16B each). ASYNCcnt-tracked.
// Chunk LDS layout: row-major 16 rows x 64 bytes.
// ---------------------------------------------------------------------------
DEV void issue_async_chunk(const __bf16* __restrict__ gbase, int mt, int ks,
                           __bf16* lds_chunk, int lane) {
#pragma unroll
    for (int i = 0; i < 2; ++i) {
        int o    = (i * 32 + lane) * 16;   // byte offset within chunk
        int row  = o >> 6;                 // 64 B per row
        int colb = o & 63;
        const char* gp = (const char*)(gbase + (size_t)(mt * 16 + row) * DD) +
                         ks * 64 + colb;
        unsigned lds = (unsigned)(size_t)((char*)lds_chunk + o);
        asm volatile("global_load_async_to_lds_b128 %0, %1, off"
                     :: "v"(lds), "v"((unsigned long long)(size_t)gp)
                     : "memory");
    }
}

// ---------------------------------------------------------------------------
// Kernel 1a: k = jobs@Wk^T, v = jobs@Wv^T ; store exp(k), exp(k)*v transposed
// as bf16 [B][D][MP] (zero-padded rows m>=M). One wave -> one 16x16 tile.
// ---------------------------------------------------------------------------
__global__ __launch_bounds__(128) void k_proj_kv(
    const float* __restrict__ jobs, const float* __restrict__ Wk,
    const float* __restrict__ Wv, __bf16* __restrict__ ekT,
    __bf16* __restrict__ ekvT) {
    int lane = threadIdx.x & 31;
    int dt = blockIdx.x * 4 + (threadIdx.x >> 5);   // D tile (0..15)
    int mt = blockIdx.y;                            // M tile (0..31)
    int b  = blockIdx.z;
    const float* A = jobs + (size_t)b * MM * DD;
    v8f ck = {}, cv = {};
#pragma unroll
    for (int ks = 0; ks < 8; ++ks) {
        v16bf a  = load_a_f32(A, mt * 16, MM - 1, DD, ks * 32, lane);
        v16bf bk = load_b_f32(Wk, dt * 16, DD, ks * 32, lane);
        v16bf bv = load_b_f32(Wv, dt * 16, DD, ks * 32, lane);
        ck = wmma_bf16(a, bk, ck);
        cv = wmma_bf16(a, bv, cv);
    }
    int d     = dt * 16 + (lane & 15);
    int mbase = mt * 16 + (lane >> 4) * 8;
    __bf16* pk = ekT  + ((size_t)b * DD + d) * MP + mbase;
    __bf16* pv = ekvT + ((size_t)b * DD + d) * MP + mbase;
#pragma unroll
    for (int i = 0; i < 8; ++i) {
        int m = mbase + i;
        float ek  = __expf(ck[i]);
        float ekv = ek * cv[i];
        bool ok = (m < MM);
        pk[i] = ok ? (__bf16)ek  : (__bf16)0.0f;
        pv[i] = ok ? (__bf16)ekv : (__bf16)0.0f;
    }
}

// ---------------------------------------------------------------------------
// Kernel 1b: q = q0@Wq0^T + q1@Wq1^T ; sigq = sigmoid(q), f32 [B][NP][D],
// zero-padded rows n>=N.
// ---------------------------------------------------------------------------
__global__ __launch_bounds__(128) void k_proj_q(
    const float* __restrict__ q0, const float* __restrict__ q1,
    const float* __restrict__ Wq0, const float* __restrict__ Wq1,
    float* __restrict__ sigq) {
    int lane = threadIdx.x & 31;
    int dt = blockIdx.x * 4 + (threadIdx.x >> 5);
    int nt = blockIdx.y;
    int b  = blockIdx.z;
    const float* A0 = q0 + (size_t)b * NN * DD;
    const float* A1 = q1 + (size_t)b * NN * DD;
    v8f c = {};
#pragma unroll
    for (int ks = 0; ks < 8; ++ks) {
        v16bf a0 = load_a_f32(A0, nt * 16, NN - 1, DD, ks * 32, lane);
        v16bf b0 = load_b_f32(Wq0, dt * 16, DD, ks * 32, lane);
        c = wmma_bf16(a0, b0, c);
        v16bf a1 = load_a_f32(A1, nt * 16, NN - 1, DD, ks * 32, lane);
        v16bf b1 = load_b_f32(Wq1, dt * 16, DD, ks * 32, lane);
        c = wmma_bf16(a1, b1, c);
    }
    int d     = dt * 16 + (lane & 15);
    int nbase = nt * 16 + (lane >> 4) * 8;
#pragma unroll
    for (int i = 0; i < 8; ++i) {
        int n = nbase + i;
        float s = (n < NN) ? (1.0f / (1.0f + __expf(-c[i]))) : 0.0f;
        sigq[((size_t)b * NP + n) * DD + d] = s;
    }
}

// ---------------------------------------------------------------------------
// Kernel 1c: jobs -> bf16, [B][MP][D], zero-padded.
// ---------------------------------------------------------------------------
__global__ __launch_bounds__(256) void k_jobs_bf16(
    const float* __restrict__ jobs, __bf16* __restrict__ jb) {
    int idx = blockIdx.x * 256 + threadIdx.x;       // B*MP*D = 2^22
    int d = idx & (DD - 1);
    int m = (idx >> 8) & (MP - 1);
    int b = idx >> 17;
    jb[idx] = (m < MM) ? (__bf16)jobs[((size_t)b * MM + m) * DD + d]
                       : (__bf16)0.0f;
}

// ---------------------------------------------------------------------------
// Kernel 2a: expb = exp(-alpha1*log_scale*cost + mask), bf16 [B][NP][MP],
// zero-padded (so padded K columns contribute nothing).
// ---------------------------------------------------------------------------
__global__ __launch_bounds__(256) void k_expbias(
    const float* __restrict__ cost, const float* __restrict__ mask,
    const float* __restrict__ alpha1, const float* __restrict__ log_scale,
    __bf16* __restrict__ expb) {
    int idx = blockIdx.x * 256 + threadIdx.x;       // B*NP*MP = 2^23
    int m = idx & (MP - 1);
    int n = (idx >> 9) & (NP - 1);
    int b = idx >> 18;
    float na1ls = -alpha1[0] * log_scale[0];
    float val = 0.0f;
    if (n < NN && m < MM) {
        size_t ci = ((size_t)b * NN + n) * MM + m;
        val = __expf(fmaf(na1ls, cost[ci], mask[ci]));
    }
    expb[idx] = (__bf16)val;
}

// ---------------------------------------------------------------------------
// Kernel 2b: num = expb@ekv, den = expb@ek (shared A fragment), weighted =
// nan_to_num(num/den), aafm = sigmoid(q)*weighted -> bf16 [B][NP][D].
// ---------------------------------------------------------------------------
__global__ __launch_bounds__(128) void k_aafm(
    const __bf16* __restrict__ expb, const __bf16* __restrict__ ekT,
    const __bf16* __restrict__ ekvT, const float* __restrict__ sigq,
    __bf16* __restrict__ aafm) {
    int lane = threadIdx.x & 31;
    int dt = blockIdx.x * 4 + (threadIdx.x >> 5);
    int nt = blockIdx.y;
    int b  = blockIdx.z;
    const __bf16* Ab = expb + (size_t)b * NP * MP;
    const __bf16* Bk = ekT  + (size_t)b * DD * MP;
    const __bf16* Bv = ekvT + (size_t)b * DD * MP;
    v8f cn = {}, cd = {};
#pragma unroll
    for (int ks = 0; ks < 16; ++ks) {               // K over MP=512
        v16bf a  = load_a_bf16(Ab, nt * 16, MP, ks * 32, lane);
        v16bf bn = load_b_bf16(Bv, dt * 16, MP, ks * 32, lane);
        v16bf bd = load_b_bf16(Bk, dt * 16, MP, ks * 32, lane);
        cn = wmma_bf16(a, bn, cn);
        cd = wmma_bf16(a, bd, cd);
    }
    int d     = dt * 16 + (lane & 15);
    int nbase = nt * 16 + (lane >> 4) * 8;
#pragma unroll
    for (int i = 0; i < 8; ++i) {
        int n = nbase + i;
        float w = cn[i] / cd[i];
        if (w != w) w = 0.0f;                               // NaN -> 0
        w = fminf(fmaxf(w, -FLT_MAX_), FLT_MAX_);           // +-Inf -> +-max
        float sq = sigq[((size_t)b * NP + n) * DD + d];
        aafm[((size_t)b * NP + n) * DD + d] = (__bf16)(sq * w);
    }
}

// ---------------------------------------------------------------------------
// Kernel 3: score = aafm@jobs^T/16 - a2*ls*cost ; 10*tanh + mask ; softmax.
// One workgroup (4 waves) per (b, 16-row N tile); 16x512 f32 LDS score tile.
// The jobs B-operand is streamed (read-once) through LDS with double-buffered
// GLOBAL_LOAD_ASYNC_TO_LDS_B128 copies, synced with s_wait_asynccnt.
// ---------------------------------------------------------------------------
__global__ __launch_bounds__(128) void k_score(
    const __bf16* __restrict__ aafm, const __bf16* __restrict__ jb,
    const float* __restrict__ cost, const float* __restrict__ mask,
    const float* __restrict__ alpha2, const float* __restrict__ log_scale,
    float* __restrict__ out) {
    __shared__ float sL[16][MP];                      // 32 KB score tile
    __shared__ float red[16][8];
    __shared__ __align__(16) __bf16 stage[4][2][16 * 32];  // 8 KB staging
    int tid  = threadIdx.x;
    int lane = tid & 31;
    int wid  = tid >> 5;
    int nt = blockIdx.x;
    int b  = blockIdx.y;
    float a2ls = alpha2[0] * log_scale[0];
    const __bf16* Ab = aafm + (size_t)b * NP * DD;
    const __bf16* Bb = jb   + (size_t)b * MP * DD;

    // Hoist A fragments (16x256 tile) into registers: reused by all 8 column
    // tiles this wave owns.
    v16bf aT[8];
#pragma unroll
    for (int ks = 0; ks < 8; ++ks)
        aT[ks] = load_a_bf16(Ab, nt * 16, DD, ks * 32, lane);

    // Prime the async pipeline with the first chunk.
    int buf = 0;
    issue_async_chunk(Bb, wid, 0, &stage[wid][0][0], lane);

    for (int mt = wid; mt < 32; mt += 4) {
        v8f c = {};
#pragma unroll
        for (int ks = 0; ks < 8; ++ks) {
            int nks = ks + 1, nmt = mt;
            if (nks == 8) { nks = 0; nmt = mt + 4; }
            bool has_next = (nmt < 32);
            if (has_next)   // prefetch next chunk into the other buffer
                issue_async_chunk(Bb, nmt, nks, &stage[wid][buf ^ 1][0], lane);
            if (has_next)
                asm volatile("s_wait_asynccnt 0x2" ::: "memory");
            else
                asm volatile("s_wait_asynccnt 0x0" ::: "memory");
            v16bf bb = load_b_bf16(&stage[wid][buf][0], 0, 32, 0, lane);
            c = wmma_bf16(aT[ks], bb, c);
            buf ^= 1;
        }
        int m   = mt * 16 + (lane & 15);
        int nlb = (lane >> 4) * 8;
#pragma unroll
        for (int i = 0; i < 8; ++i) {
            int nl = nlb + i;
            int ng = nt * 16 + nl;
            float sc;
            if (m < MM) {
                float cm = 0.0f, mk = 0.0f;
                if (ng < NN) {
                    size_t ci = ((size_t)b * NN + ng) * MM + m;
                    cm = cost[ci];
                    mk = mask[ci];
                }
                sc = 10.0f * tanhf(fmaf(c[i], 0.0625f, -a2ls * cm)) + mk;
            } else {
                sc = -__builtin_inff();              // padded cols -> exp()=0
            }
            sL[nl][m] = sc;
        }
    }
    __syncthreads();

    // Softmax over 500 valid columns: 8 threads per row.
    int row = tid >> 3;      // 0..15
    int sub = tid & 7;
    float ps = 0.0f;
    for (int m = sub; m < MM; m += 8) {
        float e = __expf(sL[row][m]);
        sL[row][m] = e;
        ps += e;
    }
    red[row][sub] = ps;
    __syncthreads();
    float tot = 0.0f;
#pragma unroll
    for (int j = 0; j < 8; ++j) tot += red[row][j];
    float inv = 1.0f / tot;
    int ng = nt * 16 + row;
    if (ng < NN) {
        float* po = out + ((size_t)b * NN + ng) * MM;
        for (int m = sub; m < MM; m += 8) po[m] = sL[row][m] * inv;
    }
}

// ---------------------------------------------------------------------------
// Launch
// ---------------------------------------------------------------------------
extern "C" void kernel_launch(void* const* d_in, const int* in_sizes, int n_in,
                              void* d_out, int out_size, void* d_ws,
                              size_t ws_size, hipStream_t stream) {
    (void)in_sizes; (void)n_in; (void)out_size; (void)ws_size;
    const float* q0        = (const float*)d_in[0];
    const float* jobs      = (const float*)d_in[1];
    const float* q1        = (const float*)d_in[2];
    const float* cost      = (const float*)d_in[3];
    const float* log_scale = (const float*)d_in[4];
    const float* mask      = (const float*)d_in[5];
    const float* Wq0       = (const float*)d_in[6];
    const float* Wq1       = (const float*)d_in[7];
    const float* Wk        = (const float*)d_in[8];
    const float* Wv        = (const float*)d_in[9];
    const float* alpha1    = (const float*)d_in[10];
    const float* alpha2    = (const float*)d_in[11];
    float* out = (float*)d_out;

    // Workspace layout (64 MiB total, all zero-padded buffers fully written
    // each call -> deterministic):
    char* ws = (char*)d_ws;
    __bf16* ekT  = (__bf16*)(ws);                       //  8 MiB [B][D][MP]
    __bf16* ekvT = (__bf16*)(ws + ((size_t)8  << 20));  //  8 MiB [B][D][MP]
    float*  sigq = (float*) (ws + ((size_t)16 << 20));  // 16 MiB [B][NP][D]
    __bf16* expb = (__bf16*)(ws + ((size_t)32 << 20));  // 16 MiB [B][NP][MP]
    __bf16* jb   = (__bf16*)(ws + ((size_t)48 << 20));  //  8 MiB [B][MP][D]
    __bf16* aafm = (__bf16*)(ws + ((size_t)56 << 20));  //  8 MiB [B][NP][D]

    k_proj_kv<<<dim3(4, 32, BB), 128, 0, stream>>>(jobs, Wk, Wv, ekT, ekvT);
    k_proj_q <<<dim3(4, 32, BB), 128, 0, stream>>>(q0, q1, Wq0, Wq1, sigq);
    k_jobs_bf16<<<(BB * MP * DD) / 256, 256, 0, stream>>>(jobs, jb);
    k_expbias<<<(BB * NP * MP) / 256, 256, 0, stream>>>(cost, mask, alpha1,
                                                        log_scale, expb);
    k_aafm<<<dim3(4, 32, BB), 128, 0, stream>>>(expb, ekT, ekvT, sigq, aafm);
    k_score<<<dim3(32, BB), 128, 0, stream>>>(aafm, jb, cost, mask, alpha2,
                                              log_scale, out);
}
